// MultiHeadAttention_71305047048985
// MI455X (gfx1250) — matrix operand order repaired and checked by reading, not compiled, not run on hardware
//
#include <hip/hip_runtime.h>
#include <hip/hip_bf16.h>

#define HEADS 8
#define EMBED 512
#define HD    64
#define NB    2
#define LSEQ  256

typedef __attribute__((ext_vector_type(16))) _Float16 v16h;
typedef __attribute__((ext_vector_type(8)))  float    v8f;

__device__ __forceinline__ v8f wmma_f16(v16h a, v16h b, v8f c) {
  // (neg_a, A, neg_b, B, c_mod, C, reuse_a, reuse_b) -> v_wmma_f32_16x16x32_f16
  return __builtin_amdgcn_wmma_f32_16x16x32_f16(false, a, false, b, (short)0, c,
                                                false, false);
}

// ---------------------------------------------------------------------------
// K1: per-head projections y = x @ W.T, output head-major [n,h,l,d] (f32)
// which: 0=v,1=k,2=q,3=pq,4=pk
// ---------------------------------------------------------------------------
__global__ void proj_kernel(const float* __restrict__ values,
                            const float* __restrict__ keys,
                            const float* __restrict__ query,
                            const float* __restrict__ pos,
                            const float* __restrict__ Wv,
                            const float* __restrict__ Wk,
                            const float* __restrict__ Wq,
                            const float* __restrict__ Wpq,
                            const float* __restrict__ Wpk,
                            float* __restrict__ vh, float* __restrict__ kh,
                            float* __restrict__ qh, float* __restrict__ pqh,
                            float* __restrict__ pkh) {
  const int NHLD = NB * HEADS * LSEQ * HD;      // 2^18
  int gid = blockIdx.x * blockDim.x + threadIdx.x;
  int which = gid >> 18;
  int r = gid & (NHLD - 1);
  int d = r & 63;
  int l = (r >> 6) & (LSEQ - 1);
  int h = (r >> 14) & (HEADS - 1);
  int n = r >> 17;
  const float* x; const float* W; float* dst;
  switch (which) {
    case 0:  x = values; W = Wv;  dst = vh;  break;
    case 1:  x = keys;   W = Wk;  dst = kh;  break;
    case 2:  x = query;  W = Wq;  dst = qh;  break;
    case 3:  x = pos;    W = Wpq; dst = pqh; break;
    default: x = pos;    W = Wpk; dst = pkh; break;
  }
  const float* xp = x + ((size_t)(n * LSEQ + l)) * EMBED + h * HD;
  const float* wp = W + d * HD;
  float acc = 0.f;
#pragma unroll 8
  for (int e = 0; e < HD; ++e) acc += xp[e] * wp[e];
  dst[r] = acc;
}

// ---------------------------------------------------------------------------
// K2: qW = qh @ Wrk, kW = kh @ Wrq   (folded rel projections; no transpose)
// ---------------------------------------------------------------------------
__global__ void relw_kernel(const float* __restrict__ qh,
                            const float* __restrict__ kh,
                            const float* __restrict__ Wrk,
                            const float* __restrict__ Wrq,
                            float* __restrict__ qW, float* __restrict__ kW) {
  const int NHLD = NB * HEADS * LSEQ * HD;
  int gid = blockIdx.x * blockDim.x + threadIdx.x;
  int which = gid >> 18;
  int r = gid & (NHLD - 1);
  int e = r & 63;
  const float* src = which ? kh : qh;
  const float* W   = which ? Wrq : Wrk;
  float* dst       = which ? kW : qW;
  const float* sp = src + (size_t)(r >> 6) * HD;
  float acc = 0.f;
#pragma unroll 8
  for (int d = 0; d < HD; ++d) acc += sp[d] * W[d * HD + e];
  dst[r] = acc;
}

// ---------------------------------------------------------------------------
// K3: energy = q.k^T + pq.pk^T   one wave per 16x16 tile, WMMA f16->f32
// A frag (16x32 f16, ISA 7.12.2): row = lane&15; K = blk*16 + (lane>>4)*8 + i
// B frag (32x16 f16): col = lane&15; K = (j>>3)*16 + (lane>>4)*8 + (j&7), same as A
// C/D (16x16 f32):    M = vgpr + 8*(lane>>4); N = lane&15
// ---------------------------------------------------------------------------
__global__ void scores_wmma_kernel(const float* __restrict__ qh,
                                   const float* __restrict__ kh,
                                   const float* __restrict__ pqh,
                                   const float* __restrict__ pkh,
                                   float* __restrict__ energy) {
  int lane = threadIdx.x;            // block = 1 wave of 32
  int qt = blockIdx.x, kt = blockIdx.y, nh = blockIdx.z;
  int row = lane & 15, kg = lane >> 4;
  const float* Q  = qh  + (size_t)nh * LSEQ * HD;
  const float* K  = kh  + (size_t)nh * LSEQ * HD;
  const float* PQ = pqh + (size_t)nh * LSEQ * HD;
  const float* PK = pkh + (size_t)nh * LSEQ * HD;

  v8f acc = {};
#pragma unroll
  for (int kc = 0; kc < 2; ++kc) {
    v16h a, b, pa, pb;
#pragma unroll
    for (int i = 0; i < 16; ++i) {
      int kk = kc * 32 + (i >> 3) * 16 + kg * 8 + (i & 7);
      a[i]  = (_Float16)Q [(qt * 16 + row) * HD + kk];
      pa[i] = (_Float16)PQ[(qt * 16 + row) * HD + kk];
    }
#pragma unroll
    for (int j = 0; j < 16; ++j) {
      int kk = kc * 32 + (j >> 3) * 16 + kg * 8 + (j & 7);
      b[j]  = (_Float16)K [(kt * 16 + row) * HD + kk];   // col index = row bits
      pb[j] = (_Float16)PK[(kt * 16 + row) * HD + kk];
    }
    acc = wmma_f16(a, b, acc);
    acc = wmma_f16(pa, pb, acc);
  }
#pragma unroll
  for (int r = 0; r < 8; ++r) {
    int q = qt * 16 + r + 8 * kg;
    energy[((size_t)nh * LSEQ + q) * LSEQ + kt * 16 + (lane & 15)] = acc[r];
  }
}

// ---------------------------------------------------------------------------
// K4: energy[n,h,q,k] += sum_e rel[n,q,k,h,e] * (qW[n,q,h,e] + kW[n,k,h,e])
// One 256-thread block per (n,q,k): 2KB coalesced rel read; wave w = head w.
// This is the HBM-roofline stage (256 MB of rel @ 23.3 TB/s ~= 11 us).
// ---------------------------------------------------------------------------
__global__ void rel_kernel(const float* __restrict__ rel,
                           const float* __restrict__ qW,
                           const float* __restrict__ kW,
                           float* __restrict__ energy) {
  int b = blockIdx.x;                 // (n*L + q)*L + k
  int n  = b >> 16;
  int qi = (b >> 8) & 255;
  int ki = b & 255;
  int t = threadIdx.x;
  int h = t >> 5;
  int lane = t & 31;
  int d = lane * 2;
  float2 rv = *(const float2*)(rel + (size_t)b * EMBED + h * HD + d);
  float2 qv = *(const float2*)(qW + (size_t)((n * HEADS + h) * LSEQ + qi) * HD + d);
  float2 kv = *(const float2*)(kW + (size_t)((n * HEADS + h) * LSEQ + ki) * HD + d);
  float p = rv.x * (qv.x + kv.x) + rv.y * (qv.y + kv.y);
#pragma unroll
  for (int off = 16; off > 0; off >>= 1) p += __shfl_xor(p, off, 32);
  if (lane == 0)
    energy[((size_t)(n * HEADS + h) * LSEQ + qi) * LSEQ + ki] += p;
}

// ---------------------------------------------------------------------------
// K5: softmax over k with 1/sqrt(2) scale + mask; emit f16 attn
// ---------------------------------------------------------------------------
__global__ void softmax_kernel(const float* __restrict__ energy,
                               const float* __restrict__ mask,
                               _Float16* __restrict__ attn) {
  __shared__ float red[256];
  int row = blockIdx.x;               // (n*H + h)*L + q,  N*H*L rows
  int t = threadIdx.x;                // k index
  int n = row >> 11;                  // H*L = 2048
  float e = energy[(size_t)row * LSEQ + t] * 0.70710678118654752f +
            (1.0f - mask[n * LSEQ + t]) * (-1e9f);
  red[t] = e; __syncthreads();
  for (int s = 128; s > 0; s >>= 1) {
    if (t < s) red[t] = fmaxf(red[t], red[t + s]);
    __syncthreads();
  }
  float m = red[0]; __syncthreads();
  float p = __expf(e - m);
  red[t] = p; __syncthreads();
  for (int s = 128; s > 0; s >>= 1) {
    if (t < s) red[t] += red[t + s];
    __syncthreads();
  }
  attn[(size_t)row * LSEQ + t] = (_Float16)(p / red[0]);
}

// ---------------------------------------------------------------------------
// K6: ctx[n,q,h*64+d] = sum_k attn[n,h,q,k] * v[n,h,k,d]  (WMMA, K=256)
// ---------------------------------------------------------------------------
__global__ void av_wmma_kernel(const _Float16* __restrict__ attn,
                               const float* __restrict__ vh,
                               float* __restrict__ ctx) {
  int lane = threadIdx.x;            // block = 1 wave
  int qt = blockIdx.x, dt = blockIdx.y, nh = blockIdx.z;
  int n = nh >> 3, h = nh & 7;
  int row = lane & 15, kg = lane >> 4;
  const _Float16* A = attn + (size_t)nh * LSEQ * LSEQ;
  const float*    V = vh   + (size_t)nh * LSEQ * HD;

  v8f acc = {};
#pragma unroll
  for (int kc = 0; kc < 8; ++kc) {
    v16h a, b;
#pragma unroll
    for (int i = 0; i < 16; ++i) {
      int kk = kc * 32 + (i >> 3) * 16 + kg * 8 + (i & 7);
      a[i] = A[(qt * 16 + row) * LSEQ + kk];
    }
#pragma unroll
    for (int j = 0; j < 16; ++j) {
      int kk = kc * 32 + (j >> 3) * 16 + kg * 8 + (j & 7);
      b[j] = (_Float16)V[kk * HD + dt * 16 + (lane & 15)];
    }
    acc = wmma_f16(a, b, acc);
  }
#pragma unroll
  for (int r = 0; r < 8; ++r) {
    int q = qt * 16 + r + 8 * kg;
    ctx[((size_t)n * LSEQ + q) * EMBED + h * HD + dt * 16 + (lane & 15)] = acc[r];
  }
}

// ---------------------------------------------------------------------------
// K7: out = ctx @ Wout.T + bout  (f32 VALU, 0.27 GFLOP)
// ---------------------------------------------------------------------------
__global__ void outproj_kernel(const float* __restrict__ ctx,
                               const float* __restrict__ Wout,
                               const float* __restrict__ bout,
                               float* __restrict__ out) {
  int gid = blockIdx.x * blockDim.x + threadIdx.x;   // (n*L+l)*512 + eo
  int nl = gid >> 9;
  int eo = gid & 511;
  const float* cp = ctx + (size_t)nl * EMBED;
  const float* wp = Wout + (size_t)eo * EMBED;
  float acc = bout[eo];
#pragma unroll 8
  for (int e = 0; e < EMBED; ++e) acc += cp[e] * wp[e];
  out[gid] = acc;
}

// ---------------------------------------------------------------------------
extern "C" void kernel_launch(void* const* d_in, const int* in_sizes, int n_in,
                              void* d_out, int out_size, void* d_ws, size_t ws_size,
                              hipStream_t stream) {
  (void)in_sizes; (void)n_in; (void)out_size; (void)ws_size;
  const float* values = (const float*)d_in[0];
  const float* keys   = (const float*)d_in[1];
  const float* query  = (const float*)d_in[2];
  const float* pos    = (const float*)d_in[3];
  const float* rel    = (const float*)d_in[4];
  const float* mask   = (const float*)d_in[5];
  const float* Wv  = (const float*)d_in[6];
  const float* Wk  = (const float*)d_in[7];
  const float* Wq  = (const float*)d_in[8];
  const float* Wpq = (const float*)d_in[9];
  const float* Wpk = (const float*)d_in[10];
  const float* Wrk = (const float*)d_in[11];
  const float* Wrq = (const float*)d_in[12];
  const float* Wout = (const float*)d_in[13];
  const float* bout = (const float*)d_in[14];
  float* out = (float*)d_out;

  const int NHLD = NB * HEADS * LSEQ * HD;     // 262144
  const int NHLL = NB * HEADS * LSEQ * LSEQ;   // 1048576
  float* ws = (float*)d_ws;
  float* qh  = ws;
  float* kh  = ws + 1 * (size_t)NHLD;
  float* vh  = ws + 2 * (size_t)NHLD;
  float* pqh = ws + 3 * (size_t)NHLD;
  float* pkh = ws + 4 * (size_t)NHLD;
  float* qW  = ws + 5 * (size_t)NHLD;
  float* kW  = ws + 6 * (size_t)NHLD;
  float* energy = ws + 7 * (size_t)NHLD;
  _Float16* attn = (_Float16*)(energy + NHLL);
  float* ctx = (float*)(attn + NHLL);

  // K1: projections (5 * 262144 outputs)
  proj_kernel<<<dim3(5 * NHLD / 256), dim3(256), 0, stream>>>(
      values, keys, query, pos, Wv, Wk, Wq, Wpq, Wpk, vh, kh, qh, pqh, pkh);
  // K2: folded rel projections (2 * 262144 outputs)
  relw_kernel<<<dim3(2 * NHLD / 256), dim3(256), 0, stream>>>(
      qh, kh, Wrk, Wrq, qW, kW);
  // K3: QK^T + pos scores via WMMA: one wave per 16x16 tile
  scores_wmma_kernel<<<dim3(LSEQ / 16, LSEQ / 16, NB * HEADS), dim3(32), 0, stream>>>(
      qh, kh, pqh, pkh, energy);
  // K4: streaming rel term (memory-roofline stage)
  rel_kernel<<<dim3(NB * LSEQ * LSEQ), dim3(256), 0, stream>>>(
      rel, qW, kW, energy);
  // K5: softmax -> f16 attn
  softmax_kernel<<<dim3(NB * HEADS * LSEQ), dim3(256), 0, stream>>>(
      energy, mask, attn);
  // K6: attn @ V via WMMA
  av_wmma_kernel<<<dim3(LSEQ / 16, HD / 16, NB * HEADS), dim3(32), 0, stream>>>(
      attn, vh, ctx);
  // K7: output projection + bias
  outproj_kernel<<<dim3(NB * LSEQ * EMBED / 256), dim3(256), 0, stream>>>(
      ctx, Wout, bout, out);
}
